// CustomRNN_56745107914878
// MI455X (gfx1250) — matrix operand order repaired and checked by reading, not compiled, not run on hardware
//
#include <hip/hip_runtime.h>
#include <hip/hip_bf16.h>

#define Bsz 64
#define Hsz 1024
#define Tsz 512

typedef __attribute__((ext_vector_type(8)))  float          v8f;
typedef __attribute__((ext_vector_type(16))) __bf16         v16bf;
typedef __attribute__((ext_vector_type(4)))  unsigned int   uint4v;
typedef __attribute__((ext_vector_type(4)))  unsigned int   v4u;
typedef __attribute__((ext_vector_type(8)))  int            v8i;
typedef __attribute__((ext_vector_type(4)))  int            v4i;

union FragBF { uint4v q[2]; v16bf v; };

// Padded LDS layout for the B operand (64 rows x 1024 bf16):
// TDM pad engine inserts 12 dwords (24 elems) after every 256 dwords (512 elems).
// Row stride = 1072 elems (2144 B): ds_load_b128 stays 16B-aligned, banks spread.
#define PAD_ELEMS(e)  ((e) + (((e) >> 9) * 24))
#define ROW_STRIDE    1072
#define LDS_BYTES     ((size_t)(Bsz * Hsz + (Bsz * Hsz / 512) * 24) * 2)   // 137216 B

__device__ __forceinline__ unsigned short f2bf(float f) {
  union { float f; unsigned u; } x; x.f = f;
  unsigned u = x.u + 0x7FFFu + ((x.u >> 16) & 1u);   // round-to-nearest-even
  return (unsigned short)(u >> 16);
}

// ---------- stage 128KB B-operand (Global -> LDS), TDM when available ----------
__device__ __forceinline__ void lds_fill_issue(unsigned short* lds, const unsigned short* gsrc) {
#if __has_builtin(__builtin_amdgcn_tensor_load_to_lds)
  if (threadIdx.x == 0) {
    const unsigned lds_off = (unsigned)(unsigned long long)(void*)lds;   // LDS aperture: low 32b = offset
    const unsigned long long ga = (unsigned long long)(const void*)gsrc;
    v4u g0;
    g0.x = 1u;                                            // count=1, user descriptor
    g0.y = lds_off;                                       // lds_addr
    g0.z = (unsigned)(ga & 0xFFFFFFFFu);                  // global_addr[31:0]
    g0.w = (unsigned)((ga >> 32) & 0x01FFFFFFu) | (2u << 30);  // global_addr[56:32] | type=2
    v8i g1;
    g1[0] = (int)((3u << 16) | (1u << 20) | (7u << 22) | (11u << 25)); // data_size=8B, pad_en, ivl=256dw, amt=12dw
    g1[1] = 0x40000000;                                   // tensor_dim0 = 16384 (8B units), bits[63:48]
    g1[2] = 0x00010000;                                   // tensor_dim0 hi=0 | tensor_dim1 = 1
    g1[3] = 0x40000000;                                   // tile_dim0 = 16384
    g1[4] = 1;                                            // tile_dim1 = 1, tile_dim2 = 0
    g1[5] = 16384;                                        // tensor_dim0_stride lo
    g1[6] = 0;                                            // stride hi / dim1_stride lo
    g1[7] = 0;                                            // dim1_stride hi
    v4i z4 = {0, 0, 0, 0};
    v8i z8 = {0, 0, 0, 0, 0, 0, 0, 0};
    __builtin_amdgcn_tensor_load_to_lds(g0, g1, z4, z4, z8, 0);
  }
#else
  const unsigned long long* s = (const unsigned long long*)gsrc;
  for (int i = threadIdx.x; i < (Bsz * Hsz) / 4; i += blockDim.x) {
    const int e = i * 4;
    *(unsigned long long*)(lds + PAD_ELEMS(e)) = s[i];    // pad-preserving copy
  }
#endif
}

__device__ __forceinline__ void lds_fill_wait() {
#if __has_builtin(__builtin_amdgcn_tensor_load_to_lds)
  if (threadIdx.x == 0) __builtin_amdgcn_s_wait_tensorcnt(0);
#endif
  __syncthreads();
}

// ---------- software-pipelined K-loop: A from global(L2), B from padded LDS ----------
__device__ __forceinline__ void kloop_lds(const unsigned short* __restrict__ aRow,
                                          const unsigned short* lds,
                                          int lm, int hi, v8f acc[4]) {
  const int kb = hi * 8;
  FragBF fa, faN, fb[4], fbN[4];
  fa.q[0] = *(const uint4v*)(aRow + hi * 8);
  fa.q[1] = *(const uint4v*)(aRow + hi * 8 + 16);
#pragma unroll
  for (int n = 0; n < 4; ++n) {
    const unsigned short* bp = lds + (n * 16 + lm) * ROW_STRIDE + kb;    // kb<512: no pad term
    fb[n].q[0] = ((const uint4v*)bp)[0];
    fb[n].q[1] = ((const uint4v*)bp)[2];
  }
  for (int k = 0; k < Hsz - 32; k += 32) {
    const int kn = k + 32;
    const int kkn = kn + kb;
    faN.q[0] = *(const uint4v*)(aRow + kn + hi * 8);                     // prefetch next A
    faN.q[1] = *(const uint4v*)(aRow + kn + hi * 8 + 16);
#pragma unroll
    for (int n = 0; n < 4; ++n) {                                       // prefetch next B
      const unsigned short* bp = lds + (n * 16 + lm) * ROW_STRIDE + kkn + ((kkn >> 9) * 24);
      fbN[n].q[0] = ((const uint4v*)bp)[0];
      fbN[n].q[1] = ((const uint4v*)bp)[2];
    }
#pragma unroll
    for (int n = 0; n < 4; ++n)
      acc[n] = __builtin_amdgcn_wmma_f32_16x16x32_bf16(false, fa.v, false, fb[n].v,
                                                       (short)0, acc[n], false, false);
    fa = faN;
#pragma unroll
    for (int n = 0; n < 4; ++n) fb[n] = fbN[n];
  }
#pragma unroll
  for (int n = 0; n < 4; ++n)
    acc[n] = __builtin_amdgcn_wmma_f32_16x16x32_bf16(false, fa.v, false, fb[n].v,
                                                     (short)0, acc[n], false, false);
}

// ---------------- weight f32 -> bf16 ----------------
__global__ void pack_weights_kernel(const float* __restrict__ src,
                                    unsigned short* __restrict__ dst, int n) {
  int i = blockIdx.x * blockDim.x + threadIdx.x;
  if (i < n) dst[i] = f2bf(src[i]);
}

// ---------------- input [B][H][T] f32 -> Xp [T][B][H] bf16 ----------------
__global__ __launch_bounds__(256) void pack_x_kernel(const float* __restrict__ in,
                                                     unsigned short* __restrict__ Xp) {
  __shared__ unsigned short tile[32][33];
  const int tx = threadIdx.x, ty = threadIdx.y;
  const int t0 = blockIdx.x * 32, h0 = blockIdx.y * 32, b = blockIdx.z;
#pragma unroll
  for (int j = 0; j < 4; ++j) {
    int h = h0 + ty + j * 8;
    tile[ty + j * 8][tx] = f2bf(in[((size_t)b * Hsz + h) * Tsz + (t0 + tx)]);
  }
  __syncthreads();
#pragma unroll
  for (int j = 0; j < 4; ++j) {
    int tt = t0 + ty + j * 8;
    Xp[((size_t)tt * Bsz + b) * Hsz + h0 + tx] = tile[tx][ty + j * 8];
  }
}

// ------- batched GEMM: Out[t][b][h] = sum_k W[h][k]*Bm[t][b][k] (+bias[h]) -------
// 8 waves/block; block TDM-stages Bm[t] (128KB) into LDS once, shared by all waves.
__global__ __launch_bounds__(256) void gemm_bt_kernel(
    const unsigned short* __restrict__ Wbf,   // [H][H] row-major bf16
    const unsigned short* __restrict__ Bm,    // [T][B][H] bf16
    const float* __restrict__ bias,           // [H] or nullptr
    float* __restrict__ Out)                  // [T][B][H] f32
{
  extern __shared__ unsigned short hbuf[];
  const int lane = threadIdx.x & 31;
  const int wave = threadIdx.x >> 5;
  const int mtile = blockIdx.x * 8 + wave;    // 0..63
  const int t = blockIdx.y;
  const int hi = lane >> 4;
  const int lm = lane & 15;

  lds_fill_issue(hbuf, Bm + (size_t)t * Bsz * Hsz);

  v8f acc[4];
  if (bias) {
    const int h0 = mtile * 16 + hi * 8;
#pragma unroll
    for (int r = 0; r < 8; ++r) {
      float bv = bias[h0 + r];
      acc[0][r] = bv; acc[1][r] = bv; acc[2][r] = bv; acc[3][r] = bv;
    }
  } else {
#pragma unroll
    for (int r = 0; r < 8; ++r) { acc[0][r] = 0.f; acc[1][r] = 0.f; acc[2][r] = 0.f; acc[3][r] = 0.f; }
  }

  lds_fill_wait();

  const unsigned short* aRow = Wbf + (size_t)(mtile * 16 + lm) * Hsz;
  kloop_lds(aRow, hbuf, lm, hi, acc);

#pragma unroll
  for (int n = 0; n < 4; ++n) {
    float* dst = Out + ((size_t)t * Bsz + (n * 16 + lm)) * Hsz + mtile * 16 + hi * 8;
    *(v8f*)dst = acc[n];
  }
}

// ---------------- persistent recurrent kernel ----------------
// 8 blocks x 8 waves = 64 waves. Per step: TDM-stage H[t-1] to LDS (overlapped with
// f32 PRE loads), WMMA K-loop, tanh -> bf16 H[t], device-wide barrier.
__global__ __launch_bounds__(256) void rnn_steps_kernel(
    const unsigned short* __restrict__ Whh,   // [H][H] bf16
    const float* __restrict__ PRE,            // [T][B][H] f32
    unsigned short* __restrict__ Hbf,         // [T][B][H] bf16
    float* __restrict__ hlast,                // [H][B] f32 (d_out tail)
    unsigned* __restrict__ syncCtr,
    int nBlocks)
{
  extern __shared__ unsigned short hbuf[];
  const int lane = threadIdx.x & 31;
  const int wave = threadIdx.x >> 5;
  const int mtile = blockIdx.x * 8 + wave;
  const int hi = lane >> 4;
  const int lm = lane & 15;
  const unsigned short* aRow = Whh + (size_t)(mtile * 16 + lm) * Hsz;

  for (int t = 0; t < Tsz; ++t) {
    if (t > 0) lds_fill_issue(hbuf, Hbf + (size_t)(t - 1) * Bsz * Hsz);

    v8f acc[4];
#pragma unroll
    for (int n = 0; n < 4; ++n) {
      const float* src = PRE + ((size_t)t * Bsz + (n * 16 + lm)) * Hsz + mtile * 16 + hi * 8;
      acc[n] = *(const v8f*)src;                       // overlaps TDM transfer
    }

    if (t > 0) {
      lds_fill_wait();
      kloop_lds(aRow, hbuf, lm, hi, acc);
    }

#pragma unroll
    for (int n = 0; n < 4; ++n) {
      union { unsigned short s[8]; uint4v q; } st;
      float hv[8];
#pragma unroll
      for (int r = 0; r < 8; ++r) { hv[r] = tanhf(acc[n][r]); st.s[r] = f2bf(hv[r]); }
      unsigned short* dst = Hbf + ((size_t)t * Bsz + (n * 16 + lm)) * Hsz + mtile * 16 + hi * 8;
      *(uint4v*)dst = st.q;
      if (t == Tsz - 1) {
        const int bcol = n * 16 + lm;
#pragma unroll
        for (int r = 0; r < 8; ++r)
          hlast[(size_t)(mtile * 16 + hi * 8 + r) * Bsz + bcol] = hv[r];
      }
    }
    // ---- device-wide step barrier ----
    __threadfence();
    __syncthreads();
    if (threadIdx.x == 0) {
      atomicAdd(syncCtr, 1u);
      const unsigned target = (unsigned)nBlocks * (unsigned)(t + 1);
      while (__hip_atomic_load(syncCtr, __ATOMIC_ACQUIRE, __HIP_MEMORY_SCOPE_AGENT) < target)
        __builtin_amdgcn_s_sleep(2);
    }
    __syncthreads();
    __threadfence();
  }
}

// ---------------- staging [T][B][H] f32 -> d_out [B][H][T] ----------------
__global__ __launch_bounds__(256) void transpose_out_kernel(const float* __restrict__ S,
                                                            float* __restrict__ out) {
  __shared__ float tile[32][33];
  const int tx = threadIdx.x, ty = threadIdx.y;
  const int t0 = blockIdx.x * 32, h0 = blockIdx.y * 32, b = blockIdx.z;
#pragma unroll
  for (int j = 0; j < 4; ++j) {
    int tt = t0 + ty + j * 8;
    tile[ty + j * 8][tx] = S[((size_t)tt * Bsz + b) * Hsz + h0 + tx];
  }
  __syncthreads();
#pragma unroll
  for (int j = 0; j < 4; ++j) {
    int h = h0 + ty + j * 8;
    out[((size_t)b * Hsz + h) * Tsz + t0 + tx] = tile[tx][ty + j * 8];
  }
}

extern "C" void kernel_launch(void* const* d_in, const int* in_sizes, int n_in,
                              void* d_out, int out_size, void* d_ws, size_t ws_size,
                              hipStream_t stream) {
  (void)in_sizes; (void)n_in; (void)out_size; (void)ws_size;
  const float* input = (const float*)d_in[0];   // [B][H][T]
  const float* W_ih  = (const float*)d_in[1];   // [H][H]
  const float* W_hh  = (const float*)d_in[2];
  const float* W_ho  = (const float*)d_in[3];
  const float* bias  = (const float*)d_in[4];   // [H]

  char* ws = (char*)d_ws;
  const size_t MB = 1024u * 1024u;
  unsigned short* Wih_bf = (unsigned short*)(ws + 0 * MB);
  unsigned short* Whh_bf = (unsigned short*)(ws + 2 * MB);
  unsigned short* Who_bf = (unsigned short*)(ws + 4 * MB);
  unsigned*       syncCtr = (unsigned*)(ws + 6 * MB);
  unsigned short* Xp  = (unsigned short*)(ws + 6 * MB + 4096);
  float*          PRE = (float*)((char*)Xp + (size_t)Tsz * Bsz * Hsz * 2);
  unsigned short* Hbf = (unsigned short*)((char*)PRE + (size_t)Tsz * Bsz * Hsz * 4);

  float* out   = (float*)d_out;
  float* hlast = out + (size_t)Bsz * Hsz * Tsz;

  (void)hipFuncSetAttribute((const void*)gemm_bt_kernel,
                            hipFuncAttributeMaxDynamicSharedMemorySize, (int)LDS_BYTES);
  (void)hipFuncSetAttribute((const void*)rnn_steps_kernel,
                            hipFuncAttributeMaxDynamicSharedMemorySize, (int)LDS_BYTES);

  (void)hipMemsetAsync(syncCtr, 0, 256, stream);

  const int nW = Hsz * Hsz;
  pack_weights_kernel<<<nW / 256, 256, 0, stream>>>(W_ih, Wih_bf, nW);
  pack_weights_kernel<<<nW / 256, 256, 0, stream>>>(W_hh, Whh_bf, nW);
  pack_weights_kernel<<<nW / 256, 256, 0, stream>>>(W_ho, Who_bf, nW);

  dim3 tb(32, 8);
  pack_x_kernel<<<dim3(Tsz / 32, Hsz / 32, Bsz), tb, 0, stream>>>(input, Xp);

  // Phase 1: PRE[t] = W_ih @ x_t + b
  gemm_bt_kernel<<<dim3(8, Tsz), 256, LDS_BYTES, stream>>>(Wih_bf, Xp, bias, PRE);

  // Phase 2: sequential recurrence (persistent, per-step grid barrier)
  rnn_steps_kernel<<<8, 256, LDS_BYTES, stream>>>(Whh_bf, PRE, Hbf, hlast, syncCtr, 8);

  // Phase 3: OUT[t] = W_ho @ h_t  (reuse PRE region as f32 staging)
  gemm_bt_kernel<<<dim3(8, Tsz), 256, LDS_BYTES, stream>>>(Who_bf, Hbf, nullptr, PRE);

  transpose_out_kernel<<<dim3(Tsz / 32, Hsz / 32, Bsz), tb, 0, stream>>>(PRE, out);
}
